// MultiHeadAttentionDebug_3264175145514
// MI455X (gfx1250) — matrix operand order repaired and checked by reading, not compiled, not run on hardware
//
#include <hip/hip_runtime.h>
#include <hip/hip_bf16.h>
#include <math.h>

typedef __bf16 bf16_t;
typedef __attribute__((ext_vector_type(8)))  __bf16 v8bf;
typedef __attribute__((ext_vector_type(16))) __bf16 v16bf;
typedef __attribute__((ext_vector_type(8)))  float  v8f;

#define B_    4
#define S_    2048
#define D_    1024
#define H_    16
#define HD_   64
#define MROWS (B_ * S_)              // 8192

// workspace element offsets (bf16 elements)
#define OFF_XB   ((size_t)0)
#define OFF_WQB  (OFF_XB  + (size_t)MROWS * D_)
#define OFF_WKB  (OFF_WQB + (size_t)D_ * D_)
#define OFF_WVB  (OFF_WKB + (size_t)D_ * D_)
#define OFF_WOB  (OFF_WVB + (size_t)D_ * D_)
#define OFF_QB   (OFF_WOB + (size_t)D_ * D_)
#define OFF_KB   (OFF_QB  + (size_t)MROWS * D_)
#define OFF_VTB  (OFF_KB  + (size_t)MROWS * D_)
#define OFF_CTX  (OFF_VTB + (size_t)MROWS * D_)
#define WS_ELEMS (OFF_CTX + (size_t)MROWS * D_)

static __device__ __forceinline__ v16bf make16(v8bf lo, v8bf hi) {
  return __builtin_shufflevector(lo, hi, 0, 1, 2, 3, 4, 5, 6, 7,
                                         8, 9, 10, 11, 12, 13, 14, 15);
}

static __device__ __forceinline__ v8f wmma_bf16(v16bf a, v16bf b, v8f c) {
  return __builtin_amdgcn_wmma_f32_16x16x32_bf16(false, a, false, b,
                                                 (short)0, c, false, false);
}

// ---------------- fp32 -> bf16 convert (8 elems / thread) -----------------
__global__ void cvt_bf16_kernel(const float4* __restrict__ src,
                                v8bf* __restrict__ dst, int n8) {
  int i = blockIdx.x * blockDim.x + threadIdx.x;
  if (i >= n8) return;
  float4 a = src[2 * i + 0];
  float4 b = src[2 * i + 1];
  v8bf o;
  o[0] = (bf16_t)a.x; o[1] = (bf16_t)a.y; o[2] = (bf16_t)a.z; o[3] = (bf16_t)a.w;
  o[4] = (bf16_t)b.x; o[5] = (bf16_t)b.y; o[6] = (bf16_t)b.z; o[7] = (bf16_t)b.w;
  dst[i] = o;
}

// ---------------- QKV projection GEMM: Y = X(M,K) * W(N,K)^T --------------
// One wave computes a 32(M) x 64(N) tile (8 wmma / k-step, B-frags reused
// across two M-subtiles). transpose_out=0 -> [BH,S,HD] (Q,K layout),
// transpose_out=1 -> [BH,HD,S] (V^T layout for PV fragments).
__global__ __launch_bounds__(256) void proj_gemm_kernel(
    const bf16_t* __restrict__ X, const bf16_t* __restrict__ W,
    bf16_t* __restrict__ Y, int transpose_out) {
  const int lane = threadIdx.x & 31;
  const int wid = blockIdx.x * 8 + (threadIdx.x >> 5);
  const int tiles_n = D_ / 64;                    // 16
  if (wid >= (MROWS / 32) * tiles_n) return;      // 4096 waves
  const int m0 = (wid / tiles_n) * 32;
  const int n0 = (wid % tiles_n) * 64;

  const int half = lane >> 4;
  const int rA   = lane & 15;
  const int kA   = half ? 8 : 0;                  // A-frag K base per lane half
  const int kB   = half ? 8 : 0;                  // B-frag K base per lane half

  v8f acc[2][4];
#pragma unroll
  for (int mt = 0; mt < 2; ++mt)
#pragma unroll
    for (int j = 0; j < 4; ++j) acc[mt][j] = v8f{};

  for (int k0 = 0; k0 < D_; k0 += 32) {
    v16bf a[2];
#pragma unroll
    for (int mt = 0; mt < 2; ++mt) {
      const bf16_t* pa = X + (size_t)(m0 + mt * 16 + rA) * D_ + k0 + kA;
      a[mt] = make16(*(const v8bf*)pa, *(const v8bf*)(pa + 16));
      if (k0 + 32 < D_) __builtin_prefetch(pa + 32, 0, 1);
    }
#pragma unroll
    for (int j = 0; j < 4; ++j) {
      const bf16_t* pb = W + (size_t)(n0 + j * 16 + rA) * D_ + k0 + kB;
      v16bf b = make16(*(const v8bf*)pb, *(const v8bf*)(pb + 16));
      acc[0][j] = wmma_bf16(a[0], b, acc[0][j]);
      acc[1][j] = wmma_bf16(a[1], b, acc[1][j]);
    }
  }
#pragma unroll
  for (int mt = 0; mt < 2; ++mt)
#pragma unroll
    for (int j = 0; j < 4; ++j) {
#pragma unroll
      for (int i = 0; i < 8; ++i) {
        int mrow = m0 + mt * 16 + i + half * 8;
        int ncol = n0 + j * 16 + rA;
        int b = mrow >> 11, s = mrow & (S_ - 1);
        int h = ncol >> 6, hd = ncol & 63;
        size_t dst = transpose_out
            ? ((size_t)(b * H_ + h) * HD_ + hd) * S_ + s
            : ((size_t)(b * H_ + h) * S_ + s) * HD_ + hd;
        Y[dst] = (bf16_t)acc[mt][j][i];
      }
    }
}

// ---------------- causal flash attention, one wave per 32-row q tile ------
// K/V fragments are loaded once per 32-k block and reused across both
// 16-row q subtiles: 16 wmma per 8KB of K/V traffic.
__global__ __launch_bounds__(256) void attn_kernel(
    const bf16_t* __restrict__ Q, const bf16_t* __restrict__ K,
    const bf16_t* __restrict__ Vt, bf16_t* __restrict__ Ctx) {
  __shared__ __align__(16) bf16_t plds[8][32 * 32];   // 2KB per-wave P bounce
  const int lane = threadIdx.x & 31;
  const int w = threadIdx.x >> 5;
  const int wid = blockIdx.x * 8 + w;
  const int qtiles = S_ / 32;                  // 64
  if (wid >= B_ * H_ * qtiles) return;         // 4096 waves
  const int bh = wid / qtiles;
  const int q0 = (wid % qtiles) * 32;

  const bf16_t* Qb = Q  + (size_t)bh * S_ * HD_;
  const bf16_t* Kb = K  + (size_t)bh * S_ * HD_;
  const bf16_t* Vb = Vt + (size_t)bh * HD_ * S_;

  const int half = lane >> 4;
  const int row  = lane & 15;
  const int kA   = half ? 8 : 0;
  const int dseg = half * 8;

  // Q fragments, resident for the whole tile: 2 subtiles x (16x64 = 2 A-frags)
  v16bf qa[2][2];
#pragma unroll
  for (int qt = 0; qt < 2; ++qt) {
    const bf16_t* pq = Qb + (size_t)(q0 + qt * 16 + row) * HD_ + kA;
    qa[qt][0] = make16(*(const v8bf*)pq,        *(const v8bf*)(pq + 16));
    qa[qt][1] = make16(*(const v8bf*)(pq + 32), *(const v8bf*)(pq + 48));
  }

  v8f acc[2][4];
  float rm[2][8], rl[2][8];
#pragma unroll
  for (int qt = 0; qt < 2; ++qt) {
#pragma unroll
    for (int j = 0; j < 4; ++j) acc[qt][j] = v8f{};
#pragma unroll
    for (int i = 0; i < 8; ++i) { rm[qt][i] = -INFINITY; rl[qt][i] = 0.0f; }
  }

  bf16_t* myLds = &plds[w][0];
  const float scale = 0.125f;                  // 1/sqrt(64)

  for (int kb = 0; kb <= q0 + 31; kb += 32) {
    // ---- scores: 2 q-subtiles x 2 k-subtiles, K-dim 64 (2 wmma each) ----
    v8f st[2][2];
#pragma unroll
    for (int qt = 0; qt < 2; ++qt) { st[qt][0] = v8f{}; st[qt][1] = v8f{}; }
#pragma unroll
    for (int t = 0; t < 2; ++t) {
      int kcol = kb + t * 16 + row;
      const bf16_t* pk = Kb + (size_t)kcol * HD_ + dseg;
      v16bf b0 = make16(*(const v8bf*)pk,        *(const v8bf*)(pk + 16));
      v16bf b1 = make16(*(const v8bf*)(pk + 32), *(const v8bf*)(pk + 48));
#pragma unroll
      for (int qt = 0; qt < 2; ++qt) {
        st[qt][t] = wmma_bf16(qa[qt][0], b0, st[qt][t]);
        st[qt][t] = wmma_bf16(qa[qt][1], b1, st[qt][t]);
      }
    }
    // ---- online softmax update per q-subtile ----
#pragma unroll
    for (int qt = 0; qt < 2; ++qt) {
      float alpha[8], p0[8], p1[8];
#pragma unroll
      for (int i = 0; i < 8; ++i) {
        int qrow = q0 + qt * 16 + i + half * 8;
        int n = lane & 15;
        float v0 = (kb + n      > qrow) ? -INFINITY : st[qt][0][i] * scale;
        float v1 = (kb + 16 + n > qrow) ? -INFINITY : st[qt][1][i] * scale;
        float mx = fmaxf(v0, v1);
#pragma unroll
        for (int msk = 1; msk < 16; msk <<= 1)
          mx = fmaxf(mx, __shfl_xor(mx, msk, 32));
        float mnew = fmaxf(rm[qt][i], mx);
        float al = __expf(rm[qt][i] - mnew);
        float e0 = __expf(v0 - mnew);
        float e1 = __expf(v1 - mnew);
        float rs = e0 + e1;
#pragma unroll
        for (int msk = 1; msk < 16; msk <<= 1) rs += __shfl_xor(rs, msk, 32);
        rl[qt][i] = rl[qt][i] * al + rs;
        rm[qt][i] = mnew;
        alpha[i] = al; p0[i] = e0; p1[i] = e1;
      }
#pragma unroll
      for (int j = 0; j < 4; ++j)
#pragma unroll
        for (int i = 0; i < 8; ++i) acc[qt][j][i] *= alpha[i];
      // C-layout P -> LDS (rows qt*16 .. qt*16+15)
#pragma unroll
      for (int i = 0; i < 8; ++i) {
        int r = qt * 16 + i + half * 8;
        int n = lane & 15;
        myLds[r * 32 + n]      = (bf16_t)p0[i];
        myLds[r * 32 + 16 + n] = (bf16_t)p1[i];
      }
    }
    asm volatile("s_wait_dscnt 0" ::: "memory");   // per-wave LDS RAW fence
    v16bf pa[2];
#pragma unroll
    for (int qt = 0; qt < 2; ++qt) {
      const bf16_t* pp = myLds + (qt * 16 + row) * 32 + kA;
      pa[qt] = make16(*(const v8bf*)pp, *(const v8bf*)(pp + 16));
    }
    // ---- acc += P(16x32) x V(32x64); V-frag reused across q subtiles ----
#pragma unroll
    for (int j = 0; j < 4; ++j) {
      const bf16_t* pv = Vb + (size_t)(j * 16 + row) * S_ + kb + dseg;
      v16bf vb = make16(*(const v8bf*)pv, *(const v8bf*)(pv + 16));
      acc[0][j] = wmma_bf16(pa[0], vb, acc[0][j]);
      acc[1][j] = wmma_bf16(pa[1], vb, acc[1][j]);
    }
  }

  // ---- normalize and write ctx (merge heads -> [B,S,D]) ----
  int b = bh >> 4, h = bh & 15;
#pragma unroll
  for (int qt = 0; qt < 2; ++qt)
#pragma unroll
    for (int j = 0; j < 4; ++j)
#pragma unroll
      for (int i = 0; i < 8; ++i) {
        int s = q0 + qt * 16 + i + half * 8;
        int d = h * 64 + j * 16 + (lane & 15);
        Ctx[(size_t)(b * S_ + s) * D_ + d] = (bf16_t)(acc[qt][j][i] / rl[qt][i]);
      }
}

// ---------------- output projection: out = ctx * Wo^T + bo (fp32) ---------
__global__ __launch_bounds__(256) void out_proj_kernel(
    const bf16_t* __restrict__ X, const bf16_t* __restrict__ W,
    const float* __restrict__ bias, float* __restrict__ out) {
  const int lane = threadIdx.x & 31;
  const int wid = blockIdx.x * 8 + (threadIdx.x >> 5);
  const int tiles_n = D_ / 64;
  if (wid >= (MROWS / 32) * tiles_n) return;
  const int m0 = (wid / tiles_n) * 32;
  const int n0 = (wid % tiles_n) * 64;

  const int half = lane >> 4;
  const int rA   = lane & 15;
  const int kA   = half ? 8 : 0;
  const int kB   = half ? 8 : 0;

  v8f acc[2][4];
#pragma unroll
  for (int mt = 0; mt < 2; ++mt)
#pragma unroll
    for (int j = 0; j < 4; ++j) acc[mt][j] = v8f{};

  for (int k0 = 0; k0 < D_; k0 += 32) {
    v16bf a[2];
#pragma unroll
    for (int mt = 0; mt < 2; ++mt) {
      const bf16_t* pa = X + (size_t)(m0 + mt * 16 + rA) * D_ + k0 + kA;
      a[mt] = make16(*(const v8bf*)pa, *(const v8bf*)(pa + 16));
      if (k0 + 32 < D_) __builtin_prefetch(pa + 32, 0, 1);
    }
#pragma unroll
    for (int j = 0; j < 4; ++j) {
      const bf16_t* pb = W + (size_t)(n0 + j * 16 + rA) * D_ + k0 + kB;
      v16bf b = make16(*(const v8bf*)pb, *(const v8bf*)(pb + 16));
      acc[0][j] = wmma_bf16(a[0], b, acc[0][j]);
      acc[1][j] = wmma_bf16(a[1], b, acc[1][j]);
    }
  }
#pragma unroll
  for (int j = 0; j < 4; ++j) {
    int ncol = n0 + j * 16 + rA;
    float bn = bias[ncol];
#pragma unroll
    for (int mt = 0; mt < 2; ++mt)
#pragma unroll
      for (int i = 0; i < 8; ++i) {
        int mrow = m0 + mt * 16 + i + half * 8;
        out[(size_t)mrow * D_ + ncol] = acc[mt][j][i] + bn;
      }
  }
}

extern "C" void kernel_launch(void* const* d_in, const int* in_sizes, int n_in,
                              void* d_out, int out_size, void* d_ws, size_t ws_size,
                              hipStream_t stream) {
  const float* x  = (const float*)d_in[0];
  const float* Wq = (const float*)d_in[1];
  const float* Wk = (const float*)d_in[2];
  const float* Wv = (const float*)d_in[3];
  const float* Wo = (const float*)d_in[4];
  const float* bo = (const float*)d_in[5];
  if (ws_size < WS_ELEMS * sizeof(bf16_t)) return;

  bf16_t* ws  = (bf16_t*)d_ws;
  bf16_t* xb  = ws + OFF_XB;
  bf16_t* wqb = ws + OFF_WQB;
  bf16_t* wkb = ws + OFF_WKB;
  bf16_t* wvb = ws + OFF_WVB;
  bf16_t* wob = ws + OFF_WOB;
  bf16_t* qb  = ws + OFF_QB;
  bf16_t* kb  = ws + OFF_KB;
  bf16_t* vtb = ws + OFF_VTB;
  bf16_t* ctx = ws + OFF_CTX;

  // fp32 -> bf16 staging
  {
    int nX8 = (MROWS * D_) / 8;           // 1,048,576
    int nW8 = (D_ * D_) / 8;              // 131,072
    cvt_bf16_kernel<<<(nX8 + 255) / 256, 256, 0, stream>>>((const float4*)x,  (v8bf*)xb,  nX8);
    cvt_bf16_kernel<<<(nW8 + 255) / 256, 256, 0, stream>>>((const float4*)Wq, (v8bf*)wqb, nW8);
    cvt_bf16_kernel<<<(nW8 + 255) / 256, 256, 0, stream>>>((const float4*)Wk, (v8bf*)wkb, nW8);
    cvt_bf16_kernel<<<(nW8 + 255) / 256, 256, 0, stream>>>((const float4*)Wv, (v8bf*)wvb, nW8);
    cvt_bf16_kernel<<<(nW8 + 255) / 256, 256, 0, stream>>>((const float4*)Wo, (v8bf*)wob, nW8);
  }

  const int gemm_waves  = (MROWS / 32) * (D_ / 64);   // 4096
  const int gemm_blocks = gemm_waves / 8;             // 512
  proj_gemm_kernel<<<gemm_blocks, 256, 0, stream>>>(xb, wqb, qb,  0);
  proj_gemm_kernel<<<gemm_blocks, 256, 0, stream>>>(xb, wkb, kb,  0);
  proj_gemm_kernel<<<gemm_blocks, 256, 0, stream>>>(xb, wvb, vtb, 1);

  const int attn_waves  = B_ * H_ * (S_ / 32);        // 4096
  const int attn_blocks = attn_waves / 8;             // 512
  attn_kernel<<<attn_blocks, 256, 0, stream>>>(qb, kb, vtb, ctx);

  out_proj_kernel<<<gemm_blocks, 256, 0, stream>>>(ctx, wob, bo, (float*)d_out);
}
